// SafeDecisionTransformer_Structure_Bayes_13134009991783
// MI455X (gfx1250) — matrix operand order repaired and checked by reading, not compiled, not run on hardware
//
#include <hip/hip_runtime.h>

// ---------------------------------------------------------------------------
// Problem constants (from reference)
// ---------------------------------------------------------------------------
#define Bn   4
#define Tn   128
#define Sn   640       // 5*T
#define BSn  2560      // B*S
#define BTn  512       // B*T
#define Hn   1024
#define NHn  16
#define Dh   64
#define Lyr  6
#define FFn  4096
#define SDn  60
#define ADn  8
#define LDn  240

typedef __attribute__((ext_vector_type(16))) __bf16 v16bf;
typedef __attribute__((ext_vector_type(8)))  float  v8f;
typedef __attribute__((ext_vector_type(4)))  int    v4i;

#define GAS __attribute__((address_space(1)))
#define LAS __attribute__((address_space(3)))

#if defined(__AMDGCN__) && __has_builtin(__builtin_amdgcn_global_load_async_to_lds_b32) && \
    __has_builtin(__builtin_amdgcn_global_load_async_to_lds_b128)
#define USE_ASYNC_LDS 1
#else
#define USE_ASYNC_LDS 0
#endif

__device__ __forceinline__ void wait_async_zero() {
#if defined(__AMDGCN__) && __has_builtin(__builtin_amdgcn_s_wait_asynccnt)
    __builtin_amdgcn_s_wait_asynccnt(0);
#else
    asm volatile("s_wait_asynccnt 0" ::: "memory");
#endif
}

__device__ __forceinline__ float gelu_exact(float x) {
    return 0.5f * x * (1.0f + erff(x * 0.70710678118654752f));
}

// ---------------------------------------------------------------------------
// Tiled bf16-WMMA GEMM: C[M,N] = act(A[M,K] @ W[K,N] + bias[N])
// A, W, bias, C f32 in global. Tiles staged into LDS as f32 via
// GLOBAL_LOAD_ASYNC_TO_LDS (DMA path, ASYNCcnt); the B tile is transposed
// to [n][k] *by the async scatter itself* (per-lane LDS dest address).
// f32 -> bf16 conversion happens during fragment construction (co-issues
// with WMMA). Block = 256 threads = 8 waves (2Mx4N); wave tile 64x32 as
// 4x2 wmma_f32_16x16x32_bf16 accumulators. BM=BN=128, KT=32.
// ---------------------------------------------------------------------------
template <int ACT> // 0=none, 1=gelu
__global__ __launch_bounds__(256)
void gemm_bias_act(const float* __restrict__ A, const float* __restrict__ W,
                   const float* __restrict__ bias, float* __restrict__ C,
                   int M, int N, int K)
{
    __shared__ float sA[128 * 32];   // [row][k]
    __shared__ float sB[128 * 32];   // [n][k]   (n-major => contiguous frag loads)

    const int t    = threadIdx.x;
    const int lane = t & 31;
    const int w    = t >> 5;
    const int wm   = w >> 2;          // 0..1
    const int wn   = w & 3;           // 0..3
    const int l15  = lane & 15;
    const int half = lane >> 4;       // 0: lanes 0-15, 1: lanes 16-31
    const int bm   = blockIdx.x;
    const int bn   = blockIdx.y;

    v8f acc[4][2];
#pragma unroll
    for (int i = 0; i < 4; ++i)
#pragma unroll
        for (int j = 0; j < 2; ++j)
#pragma unroll
            for (int e = 0; e < 8; ++e) acc[i][j][e] = 0.0f;

    const int nk = (K + 31) >> 5;
    for (int kb = 0; kb < nk; ++kb) {
        const int k0 = kb << 5;
        const bool fullTile = (k0 + 32 <= K);

        // Prefetch next K-tile (lowers to global_prefetch_b8 on gfx1250)
        if (kb + 1 < nk) {
            __builtin_prefetch(A + (size_t)(bm * 128 + (t >> 1)) * K + k0 + 32, 0, 1);
            __builtin_prefetch(W + (size_t)(k0 + 32 + (t & 31)) * N + bn * 128, 0, 1);
        }

#if USE_ASYNC_LDS
        if (fullTile) {
            // A tile 128x32 f32: 1024 x 16B chunks via async b128 DMA
#pragma unroll
            for (int e = 0; e < 4; ++e) {
                int c4 = e * 256 + t;                 // chunk id 0..1023
                int r  = c4 >> 3;
                int cc = (c4 & 7) << 2;
                __builtin_amdgcn_global_load_async_to_lds_b128(
                    (GAS v4i*)(A + (size_t)(bm * 128 + r) * K + k0 + cc),
                    (LAS v4i*)&sA[r * 32 + cc], 0, 0);
            }
            // B tile: transpose [k][n] -> [n][k] inside the async scatter
#pragma unroll
            for (int e = 0; e < 16; ++e) {
                int idx = e * 256 + t;
                int kk = idx >> 7, n = idx & 127;     // 128 consecutive n => coalesced src
                __builtin_amdgcn_global_load_async_to_lds_b32(
                    (GAS int*)(W + (size_t)(k0 + kk) * N + bn * 128 + n),
                    (LAS int*)&sB[n * 32 + kk], 0, 0);
            }
            wait_async_zero();
        } else
#endif
        {
            // Scalar fallback (K tail): stage with zero fill
#pragma unroll
            for (int e = 0; e < 16; ++e) {
                int idx = e * 256 + t;
                int r = idx >> 5, c = idx & 31;
                int gk = k0 + c;
                sA[r * 32 + c] = (gk < K) ? A[(size_t)(bm * 128 + r) * K + gk] : 0.0f;
            }
#pragma unroll
            for (int e = 0; e < 16; ++e) {
                int idx = e * 256 + t;
                int kk = idx >> 7, n = idx & 127;
                int gk = k0 + kk;
                sB[n * 32 + kk] = (gk < K) ? W[(size_t)gk * N + bn * 128 + n] : 0.0f;
            }
        }
        __syncthreads();

        // Fragments: f32 LDS -> bf16 VGPR (A layout: lanes0-15
        // K 0-15, lanes16-31 K 16-31; B: lane=col n, 16 contig k)
        v16bf af[4], bfr[2];
#pragma unroll
        for (int i = 0; i < 4; ++i) {
            int row = wm * 64 + i * 16 + l15;
            const float* p0 = &sA[row * 32 + half * 16];
            const float* p1 = &sA[row * 32 + half * 16 + 8];
#pragma unroll
            for (int e = 0; e < 8; ++e) {
                af[i][e]     = (__bf16)p0[e];
                af[i][8 + e] = (__bf16)p1[e];
            }
        }
#pragma unroll
        for (int j = 0; j < 2; ++j) {
            int n = wn * 32 + j * 16 + l15;
            const float* p = &sB[n * 32 + half * 16];
#pragma unroll
            for (int e = 0; e < 16; ++e) bfr[j][e] = (__bf16)p[e];
        }

#pragma unroll
        for (int i = 0; i < 4; ++i)
#pragma unroll
            for (int j = 0; j < 2; ++j)
                acc[i][j] = __builtin_amdgcn_wmma_f32_16x16x32_bf16(
                    false, af[i], false, bfr[j], (short)0, acc[i][j], false, false);

        __syncthreads();
    }

    // Epilogue: bias + activation, f32 store (C layout: lane->col, vgpr->row)
#pragma unroll
    for (int j = 0; j < 2; ++j) {
        int n = bn * 128 + wn * 32 + j * 16 + l15;
        float bv = bias ? bias[n] : 0.0f;
#pragma unroll
        for (int i = 0; i < 4; ++i) {
#pragma unroll
            for (int r = 0; r < 8; ++r) {
                int m = bm * 128 + wm * 64 + i * 16 + (half ? 8 + r : r);
                float v = acc[i][j][r] + bv;
                if (ACT == 1) v = gelu_exact(v);
                C[(size_t)m * N + n] = v;
            }
        }
    }
}

// ---------------------------------------------------------------------------
// LayerNorm (optionally with residual add): out = LN(x [+ res]) * g + b
// one block (256 thr) per row of H=1024
// ---------------------------------------------------------------------------
__global__ __launch_bounds__(256)
void ln_kernel(const float* __restrict__ x, const float* __restrict__ res,
               const float* __restrict__ g, const float* __restrict__ b,
               float* __restrict__ out)
{
    __shared__ float red[256];
    const int row = blockIdx.x;
    const int t   = threadIdx.x;
    const float* xr = x + (size_t)row * Hn;
    const float* rr = res ? res + (size_t)row * Hn : nullptr;

    float v[4];
    float s = 0.0f;
#pragma unroll
    for (int e = 0; e < 4; ++e) {
        int c = t + e * 256;
        float y = xr[c];
        if (rr) y += rr[c];
        v[e] = y; s += y;
    }
    red[t] = s; __syncthreads();
    for (int o = 128; o > 0; o >>= 1) { if (t < o) red[t] += red[t + o]; __syncthreads(); }
    float mu = red[0] * (1.0f / Hn);
    __syncthreads();

    float s2 = 0.0f;
#pragma unroll
    for (int e = 0; e < 4; ++e) { float d = v[e] - mu; s2 += d * d; }
    red[t] = s2; __syncthreads();
    for (int o = 128; o > 0; o >>= 1) { if (t < o) red[t] += red[t + o]; __syncthreads(); }
    float inv = rsqrtf(red[0] * (1.0f / Hn) + 1e-5f);
    __syncthreads();

    float* orow = out + (size_t)row * Hn;
#pragma unroll
    for (int e = 0; e < 4; ++e) {
        int c = t + e * 256;
        orow[c] = (v[e] - mu) * inv * g[c] + b[c];
    }
}

// ---------------------------------------------------------------------------
// Embedding + token interleave: h[b, 5t+{0..4}, :] = {ce, re, le, se, ae}
// lemb already holds gelu(lidar@Wl1+bl1)@Wl2+bl2 from the WMMA GEMMs.
// one block per (b,t) token
// ---------------------------------------------------------------------------
__global__ __launch_bounds__(256)
void embed_kernel(const int* __restrict__ ts,
                  const float* __restrict__ state, const float* __restrict__ actions,
                  const float* __restrict__ rtg, const float* __restrict__ rtgc,
                  const float* __restrict__ lemb, const float* __restrict__ emb_t,
                  const float* __restrict__ Ws, const float* __restrict__ bs,
                  const float* __restrict__ Wa, const float* __restrict__ ba,
                  const float* __restrict__ Wr, const float* __restrict__ br,
                  const float* __restrict__ Wc, const float* __restrict__ bc,
                  float* __restrict__ h)
{
    __shared__ float sst[SDn];
    __shared__ float sac[ADn];
    __shared__ float srv[2];

    const int tok = blockIdx.x;          // 0..BT-1
    const int b   = tok / Tn;
    const int tt  = tok % Tn;
    const int t   = threadIdx.x;

    if (t < SDn) sst[t] = state[(size_t)tok * SDn + t];
    if (t < ADn) sac[t] = actions[(size_t)tok * ADn + t];
    if (t == 0) { srv[0] = rtg[tok]; srv[1] = rtgc[tok]; }
    __syncthreads();

    const float* te = emb_t + (size_t)ts[tok] * Hn;
    const size_t base = ((size_t)b * Sn + 5 * tt) * (size_t)Hn;

#pragma unroll
    for (int e = 0; e < 4; ++e) {
        int c = t + e * 256;
        float tev = te[c];
        float se = bs[c];
#pragma unroll 4
        for (int j = 0; j < SDn; ++j) se += sst[j] * Ws[j * Hn + c];
        float ae = ba[c];
#pragma unroll
        for (int j = 0; j < ADn; ++j) ae += sac[j] * Wa[j * Hn + c];
        float re = br[c] + srv[0] * Wr[c];
        float ce = bc[c] + srv[1] * Wc[c];
        float le = lemb[(size_t)tok * Hn + c];
        h[base + 0 * Hn + c] = ce + tev;
        h[base + 1 * Hn + c] = re + tev;
        h[base + 2 * Hn + c] = le + tev;
        h[base + 3 * Hn + c] = se + tev;
        h[base + 4 * Hn + c] = ae + tev;
    }
}

// ---------------------------------------------------------------------------
// Causal flash attention, bf16 WMMA.
// Block = 128 threads = 4 waves; each wave handles one 16-query tile of one
// (batch, head). Grid = B * NH * (S/16/4).
// Scores: Q(16x64) @ K^T via 2 x wmma_16x16x32_bf16 per 16-key tile.
// Online softmax over C-layout (rows on VGPR+lane-half, keys on lanes);
// P restaged via LDS to A-fragment layout; P@V per 16-wide D subtile.
// ---------------------------------------------------------------------------
__global__ __launch_bounds__(128)
void attn_kernel(const float* __restrict__ q, const float* __restrict__ k,
                 const float* __restrict__ v, float* __restrict__ o)
{
    __shared__ __bf16 lK[4][32 * 64];   // [key][d]
    __shared__ __bf16 lV[4][64 * 32];   // [d][key]  (transposed for B frags)
    __shared__ __bf16 lP[4][16 * 32];   // [m][key]

    const int blk  = blockIdx.x;
    const int qg   = blk % 10;
    const int hd   = (blk / 10) % NHn;
    const int b    = blk / (10 * NHn);
    const int t    = threadIdx.x;
    const int lane = t & 31;
    const int w    = t >> 5;
    const int l15  = lane & 15;
    const int half = lane >> 4;

    __bf16* K_ = lK[w];
    __bf16* V_ = lV[w];
    __bf16* P_ = lP[w];

    const int qtile = qg * 4 + w;        // 0..39
    const int qbase = qtile * 16;
    const float scale = 0.125f;          // 1/sqrt(64)

    // Q A-fragments for both 32-wide K-dim chunks (d = 0..31, 32..63)
    v16bf qa[2];
    {
        const float* qrow = q + ((size_t)(b * Sn + qbase + l15)) * Hn + hd * Dh;
#pragma unroll
        for (int c = 0; c < 2; ++c) {
            int d0 = c * 32 + half * 16;
#pragma unroll
            for (int e = 0; e < 8; ++e) {
                qa[c][e]     = (__bf16)qrow[d0 + e];
                qa[c][8 + e] = (__bf16)qrow[d0 + 8 + e];
            }
        }
    }

    float mrow[8], lrow[8];
    v8f oacc[4];
#pragma unroll
    for (int r = 0; r < 8; ++r) { mrow[r] = -1e30f; lrow[r] = 0.0f; }
#pragma unroll
    for (int dt = 0; dt < 4; ++dt)
#pragma unroll
        for (int e = 0; e < 8; ++e) oacc[dt][e] = 0.0f;

    const int nkb = (qbase + 16 + 31) >> 5;   // causal: 32-key blocks
    for (int kb = 0; kb < nkb; ++kb) {
        // Stage 32 keys x 64 d of K and V (V transposed), f32->bf16
#pragma unroll 4
        for (int e = 0; e < 64; ++e) {
            int idx = e * 32 + lane;
            int kk = idx >> 6, d = idx & 63;
            size_t grow = ((size_t)(b * Sn + kb * 32 + kk)) * Hn + hd * Dh + d;
            K_[kk * 64 + d] = (__bf16)k[grow];
            V_[d * 32 + kk] = (__bf16)v[grow];
        }
        asm volatile("s_wait_dscnt 0" ::: "memory");   // wave-local LDS ordering

        // Scores for the two 16-key subtiles
        v8f s[2];
#pragma unroll
        for (int jt = 0; jt < 2; ++jt) {
            v16bf kf[2];
#pragma unroll
            for (int c = 0; c < 2; ++c) {
                int key = jt * 16 + l15;
                const __bf16* p = &K_[key * 64 + c * 32 + half * 16];
#pragma unroll
                for (int e = 0; e < 16; ++e) kf[c][e] = p[e];
            }
            v8f z;
#pragma unroll
            for (int e = 0; e < 8; ++e) z[e] = 0.0f;
            z = __builtin_amdgcn_wmma_f32_16x16x32_bf16(false, qa[0], false, kf[0], (short)0, z, false, false);
            z = __builtin_amdgcn_wmma_f32_16x16x32_bf16(false, qa[1], false, kf[1], (short)0, z, false, false);
            s[jt] = z;
        }

        // Scale + causal mask + online softmax (reduce across 16-lane half)
        float pm[2][8];
#pragma unroll
        for (int r = 0; r < 8; ++r) {
            int qi = qbase + (half ? 8 + r : r);
#pragma unroll
            for (int jt = 0; jt < 2; ++jt) {
                int kj = kb * 32 + jt * 16 + l15;
                float x = s[jt][r] * scale;
                pm[jt][r] = (kj > qi) ? -1e9f : x;
            }
        }
#pragma unroll
        for (int r = 0; r < 8; ++r) {
            float lm = fmaxf(pm[0][r], pm[1][r]);
#pragma unroll
            for (int m = 8; m > 0; m >>= 1) lm = fmaxf(lm, __shfl_xor(lm, m, 16));
            float mn = fmaxf(mrow[r], lm);
            float sf = __expf(mrow[r] - mn);
            float p0 = __expf(pm[0][r] - mn);
            float p1 = __expf(pm[1][r] - mn);
            float ls = p0 + p1;
#pragma unroll
            for (int m = 8; m > 0; m >>= 1) ls += __shfl_xor(ls, m, 16);
            lrow[r] = lrow[r] * sf + ls;
            mrow[r] = mn;
#pragma unroll
            for (int dt = 0; dt < 4; ++dt) oacc[dt][r] *= sf;
            pm[0][r] = p0; pm[1][r] = p1;
        }

        // Restage P (16 rows x 32 keys) as bf16 for A-fragments
#pragma unroll
        for (int r = 0; r < 8; ++r) {
            int m = half ? 8 + r : r;
            P_[m * 32 + 0 * 16 + l15] = (__bf16)pm[0][r];
            P_[m * 32 + 1 * 16 + l15] = (__bf16)pm[1][r];
        }
        asm volatile("s_wait_dscnt 0" ::: "memory");

        v16bf pa;
        {
            const __bf16* p0 = &P_[l15 * 32 + half * 16];
            const __bf16* p1 = &P_[l15 * 32 + half * 16 + 8];
#pragma unroll
            for (int e = 0; e < 8; ++e) { pa[e] = p0[e]; pa[8 + e] = p1[e]; }
        }

        // O += P @ V  (4 subtiles of 16 along D)
#pragma unroll
        for (int dt = 0; dt < 4; ++dt) {
            v16bf vf;
            const __bf16* p = &V_[(dt * 16 + l15) * 32 + half * 16];
#pragma unroll
            for (int e = 0; e < 16; ++e) vf[e] = p[e];
            oacc[dt] = __builtin_amdgcn_wmma_f32_16x16x32_bf16(
                false, pa, false, vf, (short)0, oacc[dt], false, false);
        }
    }

    // Normalize and store
#pragma unroll
    for (int dt = 0; dt < 4; ++dt)
#pragma unroll
        for (int r = 0; r < 8; ++r) {
            int m = half ? 8 + r : r;
            float val = oacc[dt][r] / lrow[r];
            o[((size_t)(b * Sn + qbase + m)) * Hn + hd * Dh + dt * 16 + l15] = val;
        }
}

// ---------------------------------------------------------------------------
// Prediction heads: one block (128 thr) per (b,t) token
// ---------------------------------------------------------------------------
__global__ __launch_bounds__(128)
void head_kernel(const float* __restrict__ h,
                 const float* __restrict__ Wps, const float* __restrict__ bps,
                 const float* __restrict__ Wpm, const float* __restrict__ bpm,
                 const float* __restrict__ Wpv, const float* __restrict__ bpv,
                 const float* __restrict__ Wpr, const float* __restrict__ bpr,
                 const float* __restrict__ Wpc, const float* __restrict__ bpc,
                 float* __restrict__ out)
{
    __shared__ float ha[Hn], hs[Hn];
    const int tok = blockIdx.x;
    const int b   = tok / Tn;
    const int tt  = tok % Tn;
    const int t   = threadIdx.x;
    const size_t base = ((size_t)b * Sn + 5 * tt) * (size_t)Hn;

    for (int c = t; c < Hn; c += 128) {
        ha[c] = h[base + 4 * Hn + c];
        hs[c] = h[base + 3 * Hn + c];
    }
    __syncthreads();

    if (t < SDn) {
        float s = bps[t];
        for (int j = 0; j < Hn; ++j) s += ha[j] * Wps[j * SDn + t];
        out[(size_t)tok * SDn + t] = s;
    } else if (t < 68) {
        int a = t - 60;
        float s = bpm[a];
        for (int j = 0; j < Hn; ++j) s += hs[j] * Wpm[j * ADn + a];
        out[30720 + (size_t)tok * ADn + a] = tanhf(s);
    } else if (t < 76) {
        int a = t - 68;
        float s = bpv[a];
        for (int j = 0; j < Hn; ++j) s += hs[j] * Wpv[j * ADn + a];
        float sg = 1.0f / (1.0f + __expf(-s));
        out[34816 + (size_t)tok * ADn + a] = 1e-6f + (1.0f - 1e-6f) * sg;
    } else if (t == 76) {
        float s = bpr[0];
        for (int j = 0; j < Hn; ++j) s += ha[j] * Wpr[j];
        out[38912 + tok] = s;
    } else if (t == 77) {
        float s = bpc[0];
        for (int j = 0; j < Hn; ++j) s += ha[j] * Wpc[j];
        out[39424 + tok] = s;
    }
}

// ---------------------------------------------------------------------------
// Host launcher
// ---------------------------------------------------------------------------
extern "C" void kernel_launch(void* const* d_in, const int* in_sizes, int n_in,
                              void* d_out, int out_size, void* d_ws, size_t ws_size,
                              hipStream_t stream)
{
    (void)in_sizes; (void)n_in; (void)out_size; (void)ws_size;

    const int*   ts      = (const int*)d_in[0];
    const float* state   = (const float*)d_in[1];
    const float* lidar   = (const float*)d_in[2];
    const float* actions = (const float*)d_in[3];
    const float* rtg     = (const float*)d_in[4];
    const float* rtgc    = (const float*)d_in[5];

    const float* emb_t = (const float*)d_in[6];
    const float* Ws    = (const float*)d_in[7];
    const float* bs    = (const float*)d_in[8];
    const float* Wa    = (const float*)d_in[9];
    const float* ba    = (const float*)d_in[10];
    const float* Wr    = (const float*)d_in[11];
    const float* br    = (const float*)d_in[12];
    const float* Wc    = (const float*)d_in[13];
    const float* bc    = (const float*)d_in[14];
    const float* Wl1   = (const float*)d_in[15];
    const float* bl1   = (const float*)d_in[16];
    const float* Wl2   = (const float*)d_in[17];
    const float* bl2   = (const float*)d_in[18];
    const float* elng  = (const float*)d_in[19];
    const float* elnb  = (const float*)d_in[20];
    const float* Wq    = (const float*)d_in[21];
    const float* bq    = (const float*)d_in[22];
    const float* Wk    = (const float*)d_in[23];
    const float* bk    = (const float*)d_in[24];
    const float* Wv    = (const float*)d_in[25];
    const float* bv    = (const float*)d_in[26];
    const float* Wo    = (const float*)d_in[27];
    const float* bo    = (const float*)d_in[28];
    const float* W1    = (const float*)d_in[29];
    const float* b1    = (const float*)d_in[30];
    const float* W2    = (const float*)d_in[31];
    const float* b2    = (const float*)d_in[32];
    const float* ln1g  = (const float*)d_in[33];
    const float* ln1b  = (const float*)d_in[34];
    const float* ln2g  = (const float*)d_in[35];
    const float* ln2b  = (const float*)d_in[36];
    const float* Wps   = (const float*)d_in[37];
    const float* bps   = (const float*)d_in[38];
    const float* Wpm   = (const float*)d_in[39];
    const float* bpm   = (const float*)d_in[40];
    const float* Wpv   = (const float*)d_in[41];
    const float* bpv   = (const float*)d_in[42];
    const float* Wpr   = (const float*)d_in[43];
    const float* bpr   = (const float*)d_in[44];
    const float* Wpc   = (const float*)d_in[45];
    const float* bpc   = (const float*)d_in[46];

    // Workspace carve-up (f32)
    float* h    = (float*)d_ws;
    float* tmp  = h    + (size_t)BSn * Hn;
    float* qb   = tmp  + (size_t)BSn * Hn;
    float* kbuf = qb   + (size_t)BSn * Hn;
    float* vbuf = kbuf + (size_t)BSn * Hn;
    float* attn = vbuf + (size_t)BSn * Hn;
    float* ffn  = attn + (size_t)BSn * Hn;
    float* lmid = ffn  + (size_t)BSn * FFn;
    float* lemb = lmid + (size_t)BTn * Hn;

    const dim3 blk256(256);
    const dim3 gLid(BTn / 128, Hn / 128);     // 4 x 8
    const dim3 gQKV(BSn / 128, Hn / 128);     // 20 x 8
    const dim3 gFF1(BSn / 128, FFn / 128);    // 20 x 32
    const dim3 gFF2(BSn / 128, Hn / 128);     // 20 x 8

    // Lidar MLP via WMMA GEMMs
    gemm_bias_act<1><<<gLid, blk256, 0, stream>>>(lidar, Wl1, bl1, lmid, BTn, Hn, LDn);
    gemm_bias_act<0><<<gLid, blk256, 0, stream>>>(lmid, Wl2, bl2, lemb, BTn, Hn, Hn);

    // Embedding + interleave + embedding LayerNorm
    embed_kernel<<<BTn, blk256, 0, stream>>>(ts, state, actions, rtg, rtgc, lemb, emb_t,
                                             Ws, bs, Wa, ba, Wr, br, Wc, bc, h);
    ln_kernel<<<BSn, blk256, 0, stream>>>(h, nullptr, elng, elnb, h);

    // Transformer layers
    for (int i = 0; i < Lyr; ++i) {
        const size_t wOff = (size_t)i * Hn * Hn;
        const size_t bOff = (size_t)i * Hn;

        gemm_bias_act<0><<<gQKV, blk256, 0, stream>>>(h, Wq + wOff, bq + bOff, qb,   BSn, Hn, Hn);
        gemm_bias_act<0><<<gQKV, blk256, 0, stream>>>(h, Wk + wOff, bk + bOff, kbuf, BSn, Hn, Hn);
        gemm_bias_act<0><<<gQKV, blk256, 0, stream>>>(h, Wv + wOff, bv + bOff, vbuf, BSn, Hn, Hn);

        attn_kernel<<<Bn * NHn * 10, 128, 0, stream>>>(qb, kbuf, vbuf, attn);

        gemm_bias_act<0><<<gQKV, blk256, 0, stream>>>(attn, Wo + wOff, bo + bOff, tmp, BSn, Hn, Hn);
        ln_kernel<<<BSn, blk256, 0, stream>>>(tmp, h, ln1g + bOff, ln1b + bOff, h);

        gemm_bias_act<1><<<gFF1, blk256, 0, stream>>>(h,   W1 + (size_t)i * Hn * FFn, b1 + (size_t)i * FFn, ffn, BSn, FFn, Hn);
        gemm_bias_act<0><<<gFF2, blk256, 0, stream>>>(ffn, W2 + (size_t)i * FFn * Hn, b2 + bOff,            tmp, BSn, Hn, FFn);
        ln_kernel<<<BSn, blk256, 0, stream>>>(tmp, h, ln2g + bOff, ln2b + bOff, h);
    }

    // Heads
    head_kernel<<<BTn, 128, 0, stream>>>(h, Wps, bps, Wpm, bpm, Wpv, bpv,
                                         Wpr, bpr, Wpc, bpc, (float*)d_out);
}
